// ESN_38577396252852
// MI455X (gfx1250) — compile-verified, matrix operand order checked by reading, not run on hardware
//
#include <hip/hip_runtime.h>
#include <math.h>

// ---------------------------------------------------------------------------
// ESN persistent-kernel implementation for gfx1250 (MI455X).
//   pre_t = u_t @ w_in^T + x_t @ w_hh^T + bias      (32x1024 per step)
//   x_{t+1} = 0.5*x_t + 0.5*tanh(pre_t),  out[b][p][t] = x_{t+1}
// Strategy: 128 waves, one 16x16 WMMA C-tile per wave, 36 bf16 WMMAs per
// step per wave. ALL 36 weight B-fragments pinned in VGPRs for the whole
// 1024-step recurrence; u A-fragments double-buffered one step ahead;
// state kept in registers; one global barrier per step.
// ---------------------------------------------------------------------------

typedef __attribute__((ext_vector_type(16))) __bf16 v16bf;
typedef __attribute__((ext_vector_type(8)))  float  v8f;
typedef __attribute__((ext_vector_type(4)))  unsigned int u32x4;

#define B_   32
#define H_   128
#define P_   1024
#define L_   1024
#define LEAK 0.5f

#define NTILES 64      // P/16 N-tiles
#define KCH_U  4       // H/32 K-chunks from u
#define KCH_X  32      // P/32 K-chunks from x
#define KCH    36      // total K-chunks (K = 1152)
#define GBLK   16      // persistent workgroups
#define WPB    8       // waves per block (256 threads, wave32)

// workspace layout (bytes)
#define WSZ_W  (NTILES * KCH * 32 * 16 * 2)        // 2,359,296  packed weights
#define WOFF_U (WSZ_W)
#define WSZ_U  (L_ * 2 * KCH_U * 32 * 16 * 2)      // 8,388,608  packed u
#define WOFF_X (WOFF_U + WSZ_U)
#define WSZ_X  (2 * B_ * P_ * 2)                   // 131,072    x bf16 double buffer
#define WOFF_C (WOFF_X + WSZ_X)                    // barrier counter

__device__ __forceinline__ unsigned short f2bf(float x) {
  unsigned u = __float_as_uint(x);
  unsigned r = u + 0x7FFFu + ((u >> 16) & 1u);     // round-to-nearest-even
  return (unsigned short)(r >> 16);
}

// (lane, j=v*2+h) -> K offset inside a 32-wide K-chunk, per the CDNA5 ISA
// 16-bit A/B fragment layout.
__device__ __forceinline__ int kmap(int lane, int j) {
  int v = j >> 1, h = j & 1;
  return 16 * (v >> 2) + 8 * (lane >> 4) + 2 * (v & 3) + h;
}

// --- pre-pack: combined [w_in ; w_hh] -> bf16 B-fragment order -------------
__global__ void esn_pack_w(const float* __restrict__ w_in,
                           const float* __restrict__ w_hh,
                           unsigned short* __restrict__ wfrag) {
  int i = blockIdx.x * blockDim.x + threadIdx.x;
  if (i >= NTILES * KCH * 32 * 16) return;
  int j    = i & 15;
  int lane = (i >> 4) & 31;
  int rem  = i >> 9;
  int ch   = rem % KCH;
  int nt   = rem / KCH;
  int K = ch * 32 + kmap(lane, j);
  int N = nt * 16 + (lane & 15);
  float v = (K < H_) ? w_in[N * H_ + K]
                     : w_hh[(size_t)N * P_ + (size_t)(K - H_)];
  wfrag[i] = f2bf(v);
}

// --- pre-pack: u (B,H,L) -> bf16 A-fragment order per timestep -------------
__global__ void esn_pack_u(const float* __restrict__ u,
                           unsigned short* __restrict__ ufrag) {
  int i = blockIdx.x * blockDim.x + threadIdx.x;
  if (i >= L_ * 2 * KCH_U * 32 * 16) return;
  int j    = i & 15;
  int lane = (i >> 4) & 31;
  int ch   = (i >> 9) & 3;
  int mt   = (i >> 11) & 1;
  int t    = i >> 12;
  int h = ch * 32 + kmap(lane, j);
  int b = mt * 16 + (lane & 15);
  ufrag[i] = f2bf(u[(size_t)b * (H_ * L_) + (size_t)h * L_ + (size_t)t]);
}

// --- init: zero x bf16 double-buffer + barrier counter ---------------------
__global__ void esn_init(unsigned short* __restrict__ xbf,
                         unsigned* __restrict__ counter) {
  int i = blockIdx.x * blockDim.x + threadIdx.x;
  if (i < 2 * B_ * P_) xbf[i] = 0;
  if (i == 0) *counter = 0u;
}

// --- persistent main kernel ------------------------------------------------
union AF { v16bf v; u32x4 q[2]; };

__global__ __launch_bounds__(256, 1) void esn_main(
    const unsigned short* __restrict__ wfrag,
    const unsigned short* __restrict__ ufrag,
    unsigned short* __restrict__ xbf,
    const float* __restrict__ bias,
    float* __restrict__ out,
    unsigned* __restrict__ counter) {
  const int lane  = threadIdx.x & 31;
  const int gw    = blockIdx.x * WPB + (threadIdx.x >> 5);   // 0..127
  const int mtile = gw & 1;                                  // B rows 0-15 / 16-31
  const int ntile = gw >> 1;                                 // 0..63
  const int m0 = mtile * 16, n0 = ntile * 16;

  const float bval = bias[n0 + (lane & 15)];                 // per-N bias
  const int   arow = m0 + (lane & 15);                       // A-fragment row
  const int   ksub = 8 * (lane >> 4);                        // A-fragment K sub-offset
  const unsigned short* wb = wfrag + ((size_t)ntile * KCH * 32 + lane) * 16;

  // ---- pin ALL 36 weight B-fragments in VGPRs for the whole recurrence ----
  AF bw[KCH];
#pragma unroll
  for (int ch = 0; ch < KCH; ++ch) {
    bw[ch].q[0] = *(const u32x4*)(wb + (size_t)ch * 512);
    bw[ch].q[1] = *(const u32x4*)(wb + (size_t)ch * 512 + 8);
  }

  // ---- u A-fragments for t=0 (double-buffered one step ahead) ----
  AF au[KCH_U];
  {
    const unsigned short* ub0 =
        ufrag + ((size_t)mtile * KCH_U * 32 + lane) * 16;
#pragma unroll
    for (int ch = 0; ch < KCH_U; ++ch) {
      au[ch].q[0] = *(const u32x4*)(ub0 + ch * 512);
      au[ch].q[1] = *(const u32x4*)(ub0 + ch * 512 + 8);
    }
  }

  // recurrent state in registers, C-tile layout: M = m0 + v + 8*(lane>=16)
  v8f xs = {0.f, 0.f, 0.f, 0.f, 0.f, 0.f, 0.f, 0.f};

  for (int t = 0; t < L_; ++t) {
    v8f c;
#pragma unroll
    for (int i = 0; i < 8; ++i) c[i] = bval;                 // seed with bias

    // ---- u GEMM: all operands already in registers ----
#pragma unroll
    for (int ch = 0; ch < KCH_U; ++ch)
      c = __builtin_amdgcn_wmma_f32_16x16x32_bf16(false, au[ch].v, false,
                                                  bw[ch].v, (short)0, c,
                                                  false, false);

    // ---- x GEMM: only A fragments come from memory (weights in regs) ----
    const unsigned short* xin =
        xbf + (size_t)(t & 1) * (B_ * P_) + (size_t)arow * P_ + ksub;
#pragma unroll
    for (int ch = 0; ch < KCH_X; ++ch) {
      AF a;
      a.q[0] = *(const u32x4*)(xin + ch * 32);
      a.q[1] = *(const u32x4*)(xin + ch * 32 + 16);
      c = __builtin_amdgcn_wmma_f32_16x16x32_bf16(false, a.v, false,
                                                  bw[KCH_U + ch].v, (short)0,
                                                  c, false, false);
    }

    // ---- load u A-fragments for t+1 (latency hides behind tail+barrier) ----
    {
      const unsigned short* ubn =
          ufrag + (((size_t)(t + 1) * 2 + mtile) * KCH_U * 32 + lane) * 16;
#pragma unroll
      for (int ch = 0; ch < KCH_U; ++ch) {
        au[ch].q[0] = *(const u32x4*)(ubn + ch * 512);
        au[ch].q[1] = *(const u32x4*)(ubn + ch * 512 + 8);
      }
    }

    // ---- leaky tanh update + publish state ----
    unsigned short* xout = xbf + (size_t)((t + 1) & 1) * (B_ * P_);
    const int Ncol = n0 + (lane & 15);
    const int rb   = m0 + 8 * (lane >> 4);
#pragma unroll
    for (int v2 = 0; v2 < 8; ++v2) {
      float xn = (1.0f - LEAK) * xs[v2] + LEAK * tanhf(c[v2]);
      xs[v2] = xn;
      int M = rb + v2;
      xout[M * P_ + Ncol] = f2bf(xn);
      out[(size_t)M * (P_ * L_) + (size_t)Ncol * L_ + (size_t)t] = xn;
    }

    // ---- device-wide barrier (monotonic target, no reset race) ----
    __threadfence();
    __syncthreads();
    if (threadIdx.x == 0) {
      atomicAdd(counter, 1u);
      const unsigned tgt = (unsigned)GBLK * (unsigned)(t + 1);
      while (__hip_atomic_load(counter, __ATOMIC_ACQUIRE,
                               __HIP_MEMORY_SCOPE_AGENT) < tgt) {
        __builtin_amdgcn_s_sleep(2);
      }
    }
    __syncthreads();
  }
}

// ---------------------------------------------------------------------------
extern "C" void kernel_launch(void* const* d_in, const int* in_sizes, int n_in,
                              void* d_out, int out_size, void* d_ws,
                              size_t ws_size, hipStream_t stream) {
  (void)in_sizes; (void)n_in; (void)out_size; (void)ws_size;
  const float* u    = (const float*)d_in[0];   // (32,128,1024)
  const float* w_in = (const float*)d_in[1];   // (1024,128)
  const float* w_hh = (const float*)d_in[2];   // (1024,1024)
  const float* bias = (const float*)d_in[3];   // (1024,)
  float* out = (float*)d_out;                  // (32,1024,1024)

  char* ws = (char*)d_ws;                      // needs ~10.4 MB
  unsigned short* wfrag = (unsigned short*)(ws);
  unsigned short* ufrag = (unsigned short*)(ws + WOFF_U);
  unsigned short* xbf   = (unsigned short*)(ws + WOFF_X);
  unsigned*       cnt   = (unsigned*)(ws + WOFF_C);

  const int nW = NTILES * KCH * 32 * 16;          // 1,179,648
  const int nU = L_ * 2 * KCH_U * 32 * 16;        // 4,194,304
  esn_pack_w<<<(nW + 255) / 256, 256, 0, stream>>>(w_in, w_hh, wfrag);
  esn_pack_u<<<(nU + 255) / 256, 256, 0, stream>>>(u, ufrag);
  esn_init<<<(2 * B_ * P_ + 255) / 256, 256, 0, stream>>>(xbf, cnt);
  esn_main<<<GBLK, WPB * 32, 0, stream>>>(wfrag, ufrag, xbf, bias, out, cnt);
}
